// SpikingEmbedding_37074157699162
// MI455X (gfx1250) — matrix operand order repaired and checked by reading, not compiled, not run on hardware
//
#include <hip/hip_runtime.h>
#include <hip/hip_bf16.h>

typedef __attribute__((ext_vector_type(16))) _Float16 v16h;
typedef __attribute__((ext_vector_type(8)))  float    v8f;

#define H 128
#define W 128
#define TSTEPS 10
#define BATCH 8
#define CIN 2
#define OC 32   // per-branch out channels; conv produces 2*OC = 64

#define WGB_N (9 * 4 * 32 * 8)   // w_gate fragments (u32 each)
#define WINB_N (4 * 32 * 8)      // w_in fragments

union PkH2 { _Float16 h[2]; unsigned u; };

// ---------------------------------------------------------------------------
// init: zero vmem (f32), spike0 (f16), output accumulator (f32)
// ---------------------------------------------------------------------------
__global__ void snn_init(float* __restrict__ vmem, _Float16* __restrict__ spk,
                         float* __restrict__ accws, long n) {
    long i = (long)blockIdx.x * blockDim.x + threadIdx.x;
    if (i < n) {
        vmem[i]  = 0.0f;
        accws[i] = 0.0f;
        spk[i]   = (_Float16)0.0f;
    }
}

// ---------------------------------------------------------------------------
// one-time weight prepack into final WMMA B-fragment order (per-lane layout):
//   B (16-bit, 32x16), contiguous k used here, A matches: lane half (l>>4) selects K base 16; VGPR v holds K=2v,2v+1
//   stored as [tap][nt][lane][v] so a lane's 8 dwords are contiguous (2x b128)
// ---------------------------------------------------------------------------
__global__ void snn_pack(const float* __restrict__ w_in, const float* __restrict__ b_in,
                         const float* __restrict__ w_gate, const float* __restrict__ b_gate,
                         unsigned* __restrict__ wgB, unsigned* __restrict__ winB,
                         float* __restrict__ bias) {
    int i = blockIdx.x * blockDim.x + threadIdx.x;

    if (i < WGB_N) {
        int v    = i & 7;
        int lane = (i >> 3) & 31;
        int nt   = (i >> 8) & 3;
        int tap  = i >> 10;
        int k0   = (lane >> 4) * 16 + 2 * v;
        int oc   = nt * 16 + (lane & 15);
        int ky   = tap / 3, kx = tap % 3;
        PkH2 pk;
        pk.h[0] = (_Float16)w_gate[(((oc * OC + k0    ) * 3) + ky) * 3 + kx];
        pk.h[1] = (_Float16)w_gate[(((oc * OC + k0 + 1) * 3) + ky) * 3 + kx];
        wgB[i] = pk.u;
    }
    if (i < WINB_N) {
        int v    = i & 7;
        int lane = (i >> 3) & 31;
        int nt   = i >> 8;
        int j    = (lane >> 4) * 8 + v;       // tap index (k0/2); kpack = tap*2 + c
        int oc   = nt * 16 + (lane & 15);
        unsigned u = 0;
        if (j < 9) {
            int ky = j / 3, kx = j % 3;
            PkH2 pk;
            pk.h[0] = (_Float16)w_in[((oc * CIN + 0) * 3 + ky) * 3 + kx];
            pk.h[1] = (_Float16)w_in[((oc * CIN + 1) * 3 + ky) * 3 + kx];
            u = pk.u;
        }
        winB[i] = u;
    }
    if (i < 64) bias[i] = b_in[i] + b_gate[i];
}

// ---------------------------------------------------------------------------
// fused step: input conv + recurrent conv (implicit GEMM via WMMA) + pointwise
// state layout: channel-last  [b][y][x][ch]  (ch = 32)
// ---------------------------------------------------------------------------
__global__ __launch_bounds__(256) void snn_step(
    const float* __restrict__ events,       // (B,T,2,H,W)
    const unsigned* __restrict__ wgB,       // prepacked w_gate fragments
    const unsigned* __restrict__ winB,      // prepacked w_in fragments
    const float* __restrict__ bias,         // b_in + b_gate (64)
    float* __restrict__ vmem,               // (B,H,W,32) f32, in-place
    const _Float16* __restrict__ spike_in,  // (B,H,W,32) f16
    _Float16* __restrict__ spike_out,       // (B,H,W,32) f16
    float* __restrict__ accws,              // (B,H,W,32) f32
    int t_src)
{
    __shared__ _Float16 lds_spk[18 * 18 * 32];   // spike halo tile, channel-last
    __shared__ unsigned lds_ev[18 * 18];         // packed (c0,c1) f16 per pixel
    __shared__ unsigned lds_wgB[WGB_N];          // [tap][nt][lane][v]
    __shared__ unsigned lds_winB[WINB_N];        // [nt][lane][v]
    __shared__ float    lds_bias[64];

    const int tid = threadIdx.x;
    const int x0 = blockIdx.x * 16;
    const int y0 = blockIdx.y * 16;
    const int b  = blockIdx.z;

    // ---- stage prepacked weight fragments: plain coalesced copies ----
    {
        const uint4* src = (const uint4*)wgB;
        uint4*       dst = (uint4*)lds_wgB;
        #pragma unroll
        for (int e = 0; e < WGB_N / 4 / 256; ++e)           // 9 iters
            dst[tid + e * 256] = src[tid + e * 256];
        if (tid < WINB_N / 4)
            ((uint4*)lds_winB)[tid] = ((const uint4*)winB)[tid];
        if (tid < 64) lds_bias[tid] = bias[tid];
    }

    // ---- stage spike halo tile (18x18 pixels x 32 ch f16 = 64B/pixel) ----
    for (int p = tid; p < 18 * 18; p += 256) {
        int py = p / 18, px = p % 18;
        int gy = y0 + py - 1, gx = x0 + px - 1;
        uint4 v0 = {0,0,0,0}, v1 = v0, v2 = v0, v3 = v0;
        if (gy >= 0 && gy < H && gx >= 0 && gx < W) {
            const uint4* src = (const uint4*)(spike_in + ((long)(b * H + gy) * W + gx) * 32);
            v0 = src[0]; v1 = src[1]; v2 = src[2]; v3 = src[3];
        }
        uint4* dst = (uint4*)(lds_spk + p * 32);
        dst[0] = v0; dst[1] = v1; dst[2] = v2; dst[3] = v3;
    }

    // ---- stage event halo tile (2 f32 ch, strided by H*W) -> packed f16 ----
    for (int p = tid; p < 18 * 18; p += 256) {
        int py = p / 18, px = p % 18;
        int gy = y0 + py - 1, gx = x0 + px - 1;
        float e0 = 0.0f, e1 = 0.0f;
        if (gy >= 0 && gy < H && gx >= 0 && gx < W) {
            long base = (((long)b * TSTEPS + t_src) * CIN) * (long)(H * W) + (long)gy * W + gx;
            e0 = events[base];
            e1 = events[base + (long)H * W];
        }
        PkH2 pk; pk.h[0] = (_Float16)e0; pk.h[1] = (_Float16)e1;
        lds_ev[p] = pk.u;
    }

    __syncthreads();

    // ---- compute: wave w owns rows 2w, 2w+1 of the 16x16 tile ----
    const int wv   = tid >> 5;
    const int lane = tid & 31;
    const int half = lane >> 4;
    const int mcol = lane & 15;

    v8f acc[2][4];
    #pragma unroll
    for (int mt = 0; mt < 2; ++mt)
        #pragma unroll
        for (int nt = 0; nt < 4; ++nt)
            acc[mt][nt] = (v8f){0.f,0.f,0.f,0.f,0.f,0.f,0.f,0.f};

    // recurrent conv: 9 taps, K=32 each
    #pragma unroll
    for (int tap = 0; tap < 9; ++tap) {
        v16h Bf[4];
        #pragma unroll
        for (int nt = 0; nt < 4; ++nt) {
            const uint4* src = (const uint4*)(lds_wgB + ((tap * 4 + nt) * 32 + lane) * 8);
            ((uint4*)&Bf[nt])[0] = src[0];
            ((uint4*)&Bf[nt])[1] = src[1];
        }
        const int dy = tap / 3, dx = tap % 3;
        #pragma unroll
        for (int mt = 0; mt < 2; ++mt) {
            const int row = wv * 2 + mt;
            const _Float16* pp = lds_spk + ((row + dy) * 18 + (mcol + dx)) * 32 + half * 16;
            v16h A;
            ((uint4*)&A)[0] = *(const uint4*)pp;          // channels half*16 .. +7
            ((uint4*)&A)[1] = *(const uint4*)(pp + 8);    // channels half*16+8 .. +15
            #pragma unroll
            for (int nt = 0; nt < 4; ++nt)
                acc[mt][nt] = __builtin_amdgcn_wmma_f32_16x16x32_f16(
                    false, A, false, Bf[nt], (short)0, acc[mt][nt], false, false);
        }
    }

    // input conv: one padded K=32 GEMM (kpack = tap*2 + c)
    {
        v16h Bf[4];
        #pragma unroll
        for (int nt = 0; nt < 4; ++nt) {
            const uint4* src = (const uint4*)(lds_winB + (nt * 32 + lane) * 8);
            ((uint4*)&Bf[nt])[0] = src[0];
            ((uint4*)&Bf[nt])[1] = src[1];
        }
        // per-lane event-tile offsets for each A VGPR (mt-invariant part)
        int evoff[8];
        #pragma unroll
        for (int v = 0; v < 8; ++v) {
            int j = half * 8 + v;    // contiguous k: kbase/2 = tap
            evoff[v] = (j < 9) ? ((j / 3) * 18 + (j % 3) + mcol) : -1;
        }
        #pragma unroll
        for (int mt = 0; mt < 2; ++mt) {
            const int rbase = (wv * 2 + mt) * 18;
            v16h A;
            unsigned* au = (unsigned*)&A;
            #pragma unroll
            for (int v = 0; v < 8; ++v)
                au[v] = (evoff[v] >= 0) ? lds_ev[rbase + evoff[v]] : 0u;
            #pragma unroll
            for (int nt = 0; nt < 4; ++nt)
                acc[mt][nt] = __builtin_amdgcn_wmma_f32_16x16x32_f16(
                    false, A, false, Bf[nt], (short)0, acc[mt][nt], false, false);
        }
    }

    // ---- epilogue: gate, integrate, spike, reset, accumulate ----
    // C/D element: reg r, lane -> pixel m = r + 8*half, channel n = mcol (+16*nt)
    #pragma unroll
    for (int mt = 0; mt < 2; ++mt) {
        const int gy = y0 + wv * 2 + mt;
        #pragma unroll
        for (int nt = 0; nt < 2; ++nt) {
            const int ch = nt * 16 + mcol;
            const float bg = lds_bias[ch];
            const float bc = lds_bias[32 + ch];
            #pragma unroll
            for (int r = 0; r < 8; ++r) {
                const int gx = x0 + r + half * 8;
                const long idx = ((long)(b * H + gy) * W + gx) * 32 + ch;
                float g    = acc[mt][nt][r]     + bg;
                float cur  = acc[mt][nt + 2][r] + bc;
                float gate = 1.0f / (1.0f + __expf(-g));
                float vm   = vmem[idx];
                float vnr  = gate * vm + cur;
                float spk  = (vnr - 0.5f) > 0.0f ? 1.0f : 0.0f;
                vmem[idx]      = vnr * (1.0f - spk);
                spike_out[idx] = (_Float16)spk;
                accws[idx]    += vnr;
            }
        }
    }
}

// ---------------------------------------------------------------------------
// final: channel-last accumulator -> NCHW output
// ---------------------------------------------------------------------------
__global__ void snn_out(const float* __restrict__ accws, float* __restrict__ out, long n) {
    long i = (long)blockIdx.x * blockDim.x + threadIdx.x;
    if (i >= n) return;
    int x  = (int)(i & 127);
    int y  = (int)((i >> 7) & 127);
    int ch = (int)((i >> 14) & 31);
    int bq = (int)(i >> 19);
    out[i] = accws[((long)(bq * H + y) * W + x) * 32 + ch];
}

extern "C" void kernel_launch(void* const* d_in, const int* in_sizes, int n_in,
                              void* d_out, int out_size, void* d_ws, size_t ws_size,
                              hipStream_t stream) {
    const float* events = (const float*)d_in[0];
    const float* w_in   = (const float*)d_in[1];
    const float* b_in   = (const float*)d_in[2];
    const float* w_gate = (const float*)d_in[3];
    const float* b_gate = (const float*)d_in[4];
    float* out = (float*)d_out;

    char* ws = (char*)d_ws;
    float*    vmem  = (float*)ws;                               // 16 MB
    _Float16* spkA  = (_Float16*)(ws + (16L << 20));            //  8 MB
    _Float16* spkB  = (_Float16*)(ws + (24L << 20));            //  8 MB
    float*    accws = (float*)(ws + (32L << 20));               // 16 MB
    unsigned* wgB   = (unsigned*)(ws + (48L << 20));            // 36 KB
    unsigned* winB  = (unsigned*)(ws + (48L << 20) + WGB_N * 4);
    float*    biasw = (float*)(ws + (48L << 20) + (WGB_N + WINB_N) * 4);

    const long n = (long)BATCH * OC * H * W;   // 4,194,304
    snn_init<<<(unsigned)((n + 255) / 256), 256, 0, stream>>>(vmem, spkA, accws, n);
    snn_pack<<<(WGB_N + 255) / 256, 256, 0, stream>>>(w_in, b_in, w_gate, b_gate,
                                                      wgB, winB, biasw);

    dim3 grid(W / 16, H / 16, BATCH);
    dim3 block(256);
    for (int t = 0; t < TSTEPS; ++t) {
        const int t_src = (TSTEPS - 1) - t;       // reversed time axis
        const _Float16* sin = (t & 1) ? spkB : spkA;
        _Float16*       sou = (t & 1) ? spkA : spkB;
        snn_step<<<grid, block, 0, stream>>>(events, wgB, winB, biasw,
                                             vmem, sin, sou, accws, t_src);
    }

    snn_out<<<(unsigned)((n + 255) / 256), 256, 0, stream>>>(accws, out, n);
}
